// RGCN_4492535792199
// MI455X (gfx1250) — matrix operand order repaired and checked by reading, not compiled, not run on hardware
//
#include <hip/hip_runtime.h>

#define RGCN_R 8

typedef float v8f __attribute__((ext_vector_type(8)));
typedef __bf16 v16bf __attribute__((ext_vector_type(16)));

// ---------------------------------------------------------------------------
// small utility kernels
// ---------------------------------------------------------------------------
__global__ void zero_i32_kernel(int* __restrict__ p, int n) {
  int i = blockIdx.x * blockDim.x + threadIdx.x;
  if (i < n) p[i] = 0;
}

__global__ void count_kernel(const int* __restrict__ dst, const int* __restrict__ et,
                             int* __restrict__ cnt, int E) {
  int e = blockIdx.x * blockDim.x + threadIdx.x;
  if (e < E) atomicAdd(&cnt[(size_t)dst[e] * RGCN_R + et[e]], 1);
}

// in-place: int count -> float 1/max(count,1)
__global__ void norm_kernel(int* __restrict__ cnt, int n) {
  int i = blockIdx.x * blockDim.x + threadIdx.x;
  if (i < n) {
    int c = cnt[i];
    float v = 1.0f / (c > 0 ? (float)c : 1.0f);
    ((float*)cnt)[i] = v;
  }
}

// ---------------------------------------------------------------------------
// WMMA fragment loaders: fp32 source -> bf16 hi + bf16 lo correction
// A: 16x32 (MxK), 16-bit layout (ISA 7.12.2):
//   lanes 0-15: M=lane, elems 0-7 -> K=0..7, elems 8-15 -> K=16..23
//   lanes 16-31: M=lane-16, elems 0-7 -> K=8..15, elems 8-15 -> K=24..31
// ---------------------------------------------------------------------------
template<bool RELU>
__device__ __forceinline__ void load_a_frag(const float* __restrict__ A, int lda, int row0,
                                            int kk, int lane, v16bf& hi, v16bf& lo) {
  const int m  = lane & 15;
  const int kb = (lane < 16) ? 0 : 8;
  const float* ap = A + (size_t)(row0 + m) * lda + kk + kb;
#pragma unroll
  for (int i = 0; i < 8; ++i) {
    float v0 = ap[i];
    float v1 = ap[16 + i];
    if (RELU) { v0 = fmaxf(v0, 0.0f); v1 = fmaxf(v1, 0.0f); }
    __bf16 h0 = (__bf16)v0;
    __bf16 h1 = (__bf16)v1;
    hi[i]     = h0;  lo[i]     = (__bf16)(v0 - (float)h0);
    hi[8 + i] = h1;  lo[8 + i] = (__bf16)(v1 - (float)h1);
  }
}

// B: 32x16 (KxN), 16-bit layout: N = lane&15;
//   lanes 0-15: elems 0-7 -> K=0..7, elems 8-15 -> K=16..23; lanes 16-31: K=8..15, K=24..31
__device__ __forceinline__ void load_b_frag(const float* __restrict__ B, int ldb, int col0,
                                            int kk, int lane, v16bf& hi, v16bf& lo) {
  const int n  = lane & 15;
  const int kb = (lane < 16) ? 0 : 8;
  const float* bp = B + (size_t)(kk + kb) * ldb + col0 + n;
#pragma unroll
  for (int i = 0; i < 16; ++i) {
    float v = bp[(size_t)(i < 8 ? i : i + 8) * ldb];
    __bf16 h = (__bf16)v;
    hi[i] = h;
    lo[i] = (__bf16)(v - (float)h);
  }
}

// ---------------------------------------------------------------------------
// Weight pre-pack: split fp32 weights once into bf16 hi/lo WMMA B fragments.
// Fragment id = (r * FT + t) * KT + kb ; storage: [frag][hi=0/lo=1][lane] v16bf.
// One wave packs one fragment.
// ---------------------------------------------------------------------------
template<int FT, int KT>
__global__ void pack_w_kernel(const float* __restrict__ W, const float* __restrict__ rootW,
                              v16bf* __restrict__ packed) {
  const int K = KT * 32;
  const int F = FT * 16;
  const int wave = threadIdx.x >> 5;
  const int lane = threadIdx.x & 31;
  const int frag = blockIdx.x * (blockDim.x >> 5) + wave;
  const int total = (RGCN_R + 1) * FT * KT;
  if (frag >= total) return;
  const int kb = frag % KT;
  const int t  = (frag / KT) % FT;
  const int r  = frag / (KT * FT);
  const float* __restrict__ B = (r == RGCN_R) ? rootW : (W + (size_t)r * K * F);
  v16bf hi, lo;
  load_b_frag(B, F, t * 16, kb * 32, lane, hi, lo);
  packed[((size_t)frag * 2 + 0) * 32 + lane] = hi;
  packed[((size_t)frag * 2 + 1) * 32 + lane] = lo;
}

// ---------------------------------------------------------------------------
// Batched GEMM: for blockIdx.y = r in [0, R): xt[r] = A @ W[r]   ([N,K]x[K,F])
//               for blockIdx.y = R:           hinit = A @ rootW + bias
// One wave computes a 16-row x F stripe (FT = F/16 column tiles), bf16x3 split
// for near-fp32 accuracy, fp32 accumulation in v_wmma_f32_16x16x32_bf16.
// B fragments come pre-packed (two 32B vector loads per fragment).
// ---------------------------------------------------------------------------
template<int FT, int KT, bool RELU_A>
__global__ void __launch_bounds__(128)
rgcn_gemm_kernel(const float* __restrict__ A, const v16bf* __restrict__ packedW,
                 const float* __restrict__ bias,
                 float* __restrict__ xt, float* __restrict__ hinit,
                 int Nrows, int rowTiles) {
  const int K = KT * 32;
  const int F = FT * 16;
  const int wave = threadIdx.x >> 5;
  const int lane = threadIdx.x & 31;
  const int rowTile = blockIdx.x * (blockDim.x >> 5) + wave;
  if (rowTile >= rowTiles) return;  // whole wave exits together: EXEC stays all-ones

  const int r = blockIdx.y;
  const bool isRoot = (r == RGCN_R);
  float* __restrict__ out = isRoot ? hinit : (xt + (size_t)r * Nrows * F);

  const int row0 = rowTile * 16;
  const int col_lane = lane & 15;

  v8f c[FT];
#pragma unroll
  for (int t = 0; t < FT; ++t) {
    float init = isRoot ? bias[t * 16 + col_lane] : 0.0f;
#pragma unroll
    for (int v = 0; v < 8; ++v) c[t][v] = init;
  }

#pragma unroll
  for (int kb = 0; kb < KT; ++kb) {
    v16bf ahi, alo;
    load_a_frag<RELU_A>(A, K, row0, kb * 32, lane, ahi, alo);
#pragma unroll
    for (int t = 0; t < FT; ++t) {
      const size_t base = ((size_t)((r * FT + t) * KT + kb) * 2) * 32 + lane;
      v16bf bhi = packedW[base];
      v16bf blo = packedW[base + 32];
      // bf16x3 split-precision: hi*hi + lo*hi + hi*lo, fp32 accumulate
      c[t] = __builtin_amdgcn_wmma_f32_16x16x32_bf16(false, ahi, false, bhi,
                                                     (short)0, c[t], false, false);
      c[t] = __builtin_amdgcn_wmma_f32_16x16x32_bf16(false, alo, false, bhi,
                                                     (short)0, c[t], false, false);
      c[t] = __builtin_amdgcn_wmma_f32_16x16x32_bf16(false, ahi, false, blo,
                                                     (short)0, c[t], false, false);
    }
  }

  // C/D layout: VGPR v -> M = v + (lane<16 ? 0 : 8), N = lane&15
  const int rbase = row0 + ((lane < 16) ? 0 : 8);
#pragma unroll
  for (int t = 0; t < FT; ++t) {
#pragma unroll
    for (int v = 0; v < 8; ++v) {
      out[(size_t)(rbase + v) * F + t * 16 + col_lane] = c[t][v];
    }
  }
}

// ---------------------------------------------------------------------------
// Edge scatter: msg = xt[etype, src] * norm[dst*R+etype]; atomicAdd into out[dst]
// F/4 lanes per edge, float4 per lane (coalesced 16B loads from the xt row).
// ---------------------------------------------------------------------------
template<int F>
__global__ void scatter_kernel(const float* __restrict__ xt, const int* __restrict__ src,
                               const int* __restrict__ dst, const int* __restrict__ et,
                               const float* __restrict__ norm, float* __restrict__ out,
                               int E, int Nrows) {
  const int LPE = F / 4;  // lanes per edge
  long long t = (long long)blockIdx.x * blockDim.x + threadIdx.x;
  int e = (int)(t / LPE);
  int l = (int)(t % LPE);
  if (e >= E) return;
  int s = src[e], d = dst[e], r = et[e];
  float nm = norm[(size_t)d * RGCN_R + r];
  const float4 v = ((const float4*)(xt + ((size_t)r * Nrows + s) * F))[l];
  float* op = out + (size_t)d * F + l * 4;
  atomicAdd(op + 0, v.x * nm);
  atomicAdd(op + 1, v.y * nm);
  atomicAdd(op + 2, v.z * nm);
  atomicAdd(op + 3, v.w * nm);
}

// ---------------------------------------------------------------------------
// log_softmax over 32 features: one wave32 per node, cross-lane reductions
// ---------------------------------------------------------------------------
__global__ void logsoftmax32_kernel(const float* __restrict__ h2, float* __restrict__ logp,
                                    int Nrows) {
  int wave = threadIdx.x >> 5;
  int lane = threadIdx.x & 31;
  int n = blockIdx.x * (blockDim.x >> 5) + wave;
  if (n >= Nrows) return;
  float v = h2[(size_t)n * 32 + lane];
  float m = v;
#pragma unroll
  for (int off = 16; off > 0; off >>= 1) m = fmaxf(m, __shfl_xor(m, off, 32));
  float ex = __expf(v - m);
  float s = ex;
#pragma unroll
  for (int off = 16; off > 0; off >>= 1) s += __shfl_xor(s, off, 32);
  logp[(size_t)n * 32 + lane] = (v - m) - __logf(s);
}

// ---------------------------------------------------------------------------
// launch
// ---------------------------------------------------------------------------
extern "C" void kernel_launch(void* const* d_in, const int* in_sizes, int n_in,
                              void* d_out, int out_size, void* d_ws, size_t ws_size,
                              hipStream_t stream) {
  const float* x     = (const float*)d_in[0];
  const int*   eidx  = (const int*)d_in[1];   // [2, E]
  const int*   etype = (const int*)d_in[2];   // [E]
  const float* w0    = (const float*)d_in[3]; // [R,128,128]
  const float* root0 = (const float*)d_in[4]; // [128,128]
  const float* b0    = (const float*)d_in[5]; // [128]
  const float* w1    = (const float*)d_in[6]; // [R,128,32]
  const float* root1 = (const float*)d_in[7]; // [128,32]
  const float* b1    = (const float*)d_in[8]; // [32]

  const int F_IN = 128, F_HID = 128, F_OUT = 32;
  const int N = in_sizes[0] / F_IN;
  const int E = in_sizes[2];
  const int* src = eidx;
  const int* dst = eidx + E;

  // outputs: (logp [N,32], h1 [N,128], h2 [N,32]) concatenated
  float* logp = (float*)d_out;
  float* h1   = logp + (size_t)N * F_OUT;
  float* h2   = h1 + (size_t)N * F_HID;

  // workspace layout (floats):
  //   [0, NR)                       : int counts, then float norm (in place)
  //   [NR, NR + R*N*128)            : xt buffer (layer 0; reused for layer 1)
  //   then packed weights (v16bf), 32B-aligned
  const int NR = N * RGCN_R;
  int*   cnt  = (int*)d_ws;
  float* norm = (float*)d_ws;
  float* xt   = (float*)d_ws + (size_t)NR;
  const size_t xt_elems = (size_t)RGCN_R * N * F_HID;
  v16bf* pack0 = (v16bf*)(xt + xt_elems);
  const int frags0 = (RGCN_R + 1) * 8 * 4;  // FT=8, KT=4
  v16bf* pack1 = pack0 + (size_t)frags0 * 2 * 32;
  const int frags1 = (RGCN_R + 1) * 2 * 4;  // FT=2, KT=4

  // 1) per-(dst, relation) mean normalization factors
  zero_i32_kernel<<<(NR + 255) / 256, 256, 0, stream>>>(cnt, NR);
  count_kernel<<<(E + 255) / 256, 256, 0, stream>>>(dst, etype, cnt, E);
  norm_kernel<<<(NR + 255) / 256, 256, 0, stream>>>(cnt, NR);

  // 2) pre-pack both layers' weights into bf16 hi/lo WMMA fragments (once)
  pack_w_kernel<8, 4><<<(frags0 + 3) / 4, 128, 0, stream>>>(w0, root0, pack0);
  pack_w_kernel<2, 4><<<(frags1 + 3) / 4, 128, 0, stream>>>(w1, root1, pack1);

  const int rowTiles = (N + 15) / 16;           // N=50000 -> 3125 exact tiles
  dim3 gemm_grid((rowTiles + 3) / 4, RGCN_R + 1);

  // 3) layer 0: xt0[r] = x @ w0[r]; h1 = x @ root0 + b0
  rgcn_gemm_kernel<8, 4, false><<<gemm_grid, 128, 0, stream>>>(
      x, pack0, b0, xt, h1, N, rowTiles);

  // 4) layer 0 scatter: h1 += mean-normalized messages
  {
    long long threads = (long long)E * (F_HID / 4);
    scatter_kernel<128><<<(unsigned)((threads + 255) / 256), 256, 0, stream>>>(
        xt, src, dst, etype, norm, h1, E, N);
  }

  // 5) layer 1 (relu applied on load of h1): xt1[r] = relu(h1) @ w1[r];
  //    h2 = relu(h1) @ root1 + b1   (xt buffer reused)
  rgcn_gemm_kernel<2, 4, true><<<gemm_grid, 128, 0, stream>>>(
      h1, pack1, b1, xt, h2, N, rowTiles);

  // 6) layer 1 scatter
  {
    long long threads = (long long)E * (F_OUT / 4);
    scatter_kernel<32><<<(unsigned)((threads + 255) / 256), 256, 0, stream>>>(
        xt, src, dst, etype, norm, h2, E, N);
  }

  // 7) logp = log_softmax(h2, axis=1)
  logsoftmax32_kernel<<<(N + 7) / 8, 256, 0, stream>>>(h2, logp, N);
}
